// PyramidSTM_60103772340233
// MI455X (gfx1250) — matrix operand order repaired and checked by reading, not compiled, not run on hardware
//
#include <hip/hip_runtime.h>
#include <math.h>

// Problem constants (match reference)
#define BDIM 16
#define CDIM 192
#define HDIM 128
#define WDIM 128
#define NFEAT 192
#define G4 (4 * NFEAT)   // 768 gate rows
#define KU (3 * NFEAT)   // 576 im2col K for recurrent conv

typedef __attribute__((ext_vector_type(16))) __bf16        v16bf;
typedef __attribute__((ext_vector_type(8)))  float         v8f;
typedef __attribute__((ext_vector_type(4)))  unsigned int  v4u;

struct F2 { v4u lo, hi; };

static __device__ __forceinline__ unsigned short f2bf(float f) {
    unsigned int u = __float_as_uint(f);
    unsigned int r = u + 0x7FFFu + ((u >> 16) & 1u);   // round-to-nearest-even
    return (unsigned short)(r >> 16);
}
static __device__ __forceinline__ unsigned pack2bf(float lo, float hi) {
    return (unsigned)f2bf(lo) | ((unsigned)f2bf(hi) << 16);
}
static __device__ __forceinline__ float bf2f(unsigned short s) {
    return __uint_as_float(((unsigned int)s) << 16);
}
// Build a 16xbf16 WMMA fragment from two 16-byte LDS chunks.
static __device__ __forceinline__ v16bf frag2(const unsigned short* p0,
                                              const unsigned short* p1) {
    F2 f;
    f.lo = *(const v4u*)p0;
    f.hi = *(const v4u*)(p1);
    return __builtin_bit_cast(v16bf, f);
}
static __device__ __forceinline__ float hsig(float x) {
    return fminf(fmaxf(0.2f * x + 0.5f, 0.0f), 1.0f);
}

// Gate-row permutation: permuted row p (block bx owns p in [bx*256, bx*256+256))
//   p = bx*256 + s*64 + fl   <->   original gate row o = s*192 + bx*64 + fl
// so one 256-row block holds gates s=0..3 (o,f,i,g) for features bx*64..+63.
static __device__ __forceinline__ int perm_to_orig(int p) {
    int bx = p >> 8, s = (p >> 6) & 3, fl = p & 63;
    return s * NFEAT + bx * 64 + fl;
}

// ---------------------------------------------------------------- prep kernels
__global__ void prep_w_kernel(const float* __restrict__ Wiof,
                              unsigned short* __restrict__ WmP) {
    int idx = blockIdx.x * blockDim.x + threadIdx.x;
    if (idx >= G4 * CDIM) return;
    int p = idx / CDIM, c = idx % CDIM;
    int o = perm_to_orig(p);
    // pyramid mask: rows [0,64) zero cols >=64 ; rows [64,128) zero cols >=128
    float m = 1.0f;
    if (o < 64 && c >= 64) m = 0.0f;
    if (o >= 64 && o < 128 && c >= 128) m = 0.0f;
    WmP[idx] = f2bf(Wiof[(size_t)o * CDIM + c] * m);  // W_iof (768,192,1,1)
}

__global__ void prep_u_kernel(const float* __restrict__ Uiof,
                              unsigned short* __restrict__ UpkP) {
    int idx = blockIdx.x * blockDim.x + threadIdx.x;
    if (idx >= G4 * KU) return;
    int p = idx / KU, k = idx % KU;
    int o = perm_to_orig(p);
    int kw = k / NFEAT, c = k % NFEAT;
    // U_iof is (768,192,3,1): flat ((o*192 + c)*3 + kw)
    UpkP[idx] = f2bf(Uiof[((size_t)o * NFEAT + c) * 3 + kw]);
}

__global__ void prep_c_kernel(const float* __restrict__ init_c,
                              float* __restrict__ cst) {
    int idx = blockIdx.x * blockDim.x + threadIdx.x;
    if (idx >= BDIM * NFEAT * WDIM) return;
    cst[idx] = init_c[idx % (NFEAT * WDIM)];
}

// ------------------------------------------------- input-to-state batched GEMM
// grid = (3, H, B), block = 256 (8 waves). Block computes 256(M) x 128(W) of
// permuted i2s[b,h] = WmP(768x192) @ X(192x128), stored bf16 (+bias).
// Pipelined: globals for chunk k+1 in regs while chunk k computes; 2 A tiles
// per wave so each LDS B-fragment feeds two WMMAs.
__global__ __launch_bounds__(256) void i2s_gemm_kernel(
    const float* __restrict__ x, const unsigned short* __restrict__ WmP,
    const float* __restrict__ bias, unsigned short* __restrict__ i2s) {
    __shared__ __align__(16) unsigned short At[2][256][32];  // M x K
    __shared__ __align__(16) unsigned short Bt[2][128][32];  // N x K (B^T)

    const int tid = threadIdx.x;
    const int lane = tid & 31, wid = tid >> 5;
    const int Mbase = blockIdx.x * 256;
    const int h = blockIdx.y, b = blockIdx.z;

    // k0-independent staging maps
    const int am0  = tid >> 1;                 // A: 4 chunks of 8 per thread
    const int ak0  = (tid & 1) * 16;
    const int bw   = tid & 127;                // B: 8 pairs per thread
    const int bkp0 = (tid >> 7) * 8;

    v8f acc[16] = {};
    const int mo   = wid * 16;
    const int lrow = lane & 15;
    const int kgA  = (lane >> 4) * 8;
    const int ksB  = (lane >> 4) * 8;

    const int NK = CDIM / 32;   // 6 chunks

    v4u   ar[4];
    float bx[16];

    auto loadA = [&](int k0) {
        #pragma unroll
        for (int t = 0; t < 2; ++t) {
            const unsigned short* ap =
                &WmP[(size_t)(Mbase + am0 + t * 128) * CDIM + k0 + ak0];
            ar[2 * t]     = *(const v4u*)ap;
            ar[2 * t + 1] = *(const v4u*)(ap + 8);
        }
    };
    auto loadB = [&](int k0) {
        #pragma unroll
        for (int t = 0; t < 8; ++t) {
            int kc = k0 + (bkp0 + t) * 2;
            const float* xp = &x[(((size_t)b * CDIM + kc) * HDIM + h) * WDIM + bw];
            bx[2 * t]     = xp[0];
            bx[2 * t + 1] = xp[(size_t)HDIM * WDIM];
        }
    };
    auto drain = [&](int buf) {
        #pragma unroll
        for (int t = 0; t < 2; ++t) {
            *(v4u*)&At[buf][am0 + t * 128][ak0]     = ar[2 * t];
            *(v4u*)&At[buf][am0 + t * 128][ak0 + 8] = ar[2 * t + 1];
        }
        #pragma unroll
        for (int t = 0; t < 8; ++t)
            *(unsigned*)&Bt[buf][bw][(bkp0 + t) * 2] =
                pack2bf(bx[2 * t], bx[2 * t + 1]);
    };

    loadA(0); loadB(0); drain(0);
    __syncthreads();

    for (int chunk = 0; chunk < NK; ++chunk) {
        const int cur = chunk & 1;
        if (chunk + 1 < NK) {
            loadA((chunk + 1) * 32);
            loadB((chunk + 1) * 32);
            if (chunk + 2 < NK)
                __builtin_prefetch(
                    &WmP[(size_t)(Mbase + am0) * CDIM + (chunk + 2) * 32 + ak0],
                    0, 3);
        }
        v16bf a0 = frag2(&At[cur][mo + lrow][kgA], &At[cur][mo + lrow][kgA + 16]);
        v16bf a1 = frag2(&At[cur][128 + mo + lrow][kgA],
                         &At[cur][128 + mo + lrow][kgA + 16]);
        #pragma unroll
        for (int nt = 0; nt < 8; ++nt) {
            const unsigned short* bp = &Bt[cur][nt * 16 + lrow][ksB];
            v16bf bf = frag2(bp, bp + 16);
            acc[nt] = __builtin_amdgcn_wmma_f32_16x16x32_bf16(
                false, a0, false, bf, (short)0, acc[nt], false, false);
            acc[8 + nt] = __builtin_amdgcn_wmma_f32_16x16x32_bf16(
                false, a1, false, bf, (short)0, acc[8 + nt], false, false);
        }
        if (chunk + 1 < NK) drain(cur ^ 1);
        __syncthreads();
    }

    const int hi = lane >> 4, col = lane & 15;
    #pragma unroll
    for (int half = 0; half < 2; ++half) {
        #pragma unroll
        for (int nt = 0; nt < 8; ++nt) {
            #pragma unroll
            for (int r = 0; r < 8; ++r) {
                int p = Mbase + half * 128 + mo + hi * 8 + r;  // permuted row
                int n = nt * 16 + col;
                float v = acc[half * 8 + nt][r] + bias[perm_to_orig(p)];
                i2s[(((size_t)b * HDIM + h) * G4 + p) * WDIM + n] = f2bf(v);
            }
        }
    }
}

// --------------------------------------- fused recurrent step kernel (per h)
// grid = (3, B), block = 256. Block bx computes permuted gate rows
// [bx*256, bx*256+256) = all four gates for features bx*64..+63, then does the
// LSTM pointwise update in-kernel via an LDS gate buffer (no global gates).
__global__ __launch_bounds__(256) void rec_step_kernel(
    const unsigned short* __restrict__ UpkP,
    const unsigned short* __restrict__ i2s,
    float* __restrict__ out_hs,           // h history lives in d_out
    const float* __restrict__ init_h,
    float* __restrict__ cst, int h) {
    __shared__ __align__(16) unsigned short At[2][256][32];
    __shared__ __align__(16) unsigned short Bt[2][128][32];
    __shared__ __align__(16) unsigned short gL[256][128];   // bf16 gates

    const int tid = threadIdx.x;
    const int lane = tid & 31, wid = tid >> 5;
    const int Mbase = blockIdx.x * 256;
    const int b = blockIdx.y;

    const int am0  = tid >> 1;
    const int ak0  = (tid & 1) * 16;
    const int bw   = tid & 127;
    const int bkp0 = (tid >> 7) * 8;

    v8f acc[16] = {};
    const int mo   = wid * 16;
    const int lrow = lane & 15;
    const int kgA  = (lane >> 4) * 8;
    const int ksB  = (lane >> 4) * 8;

    const int NK = KU / 32;   // 18 chunks
    const float* hsrc = (h == 0) ? init_h : &out_hs[(size_t)(h - 1) * WDIM];
    const size_t cstr = (h == 0) ? (size_t)WDIM : (size_t)HDIM * WDIM;
    const size_t bstr = (h == 0) ? 0 : (size_t)NFEAT * HDIM * WDIM;

    v4u   ar[4];
    float bx[16];

    auto loadA = [&](int k0) {
        #pragma unroll
        for (int t = 0; t < 2; ++t) {
            const unsigned short* ap =
                &UpkP[(size_t)(Mbase + am0 + t * 128) * KU + k0 + ak0];
            ar[2 * t]     = *(const v4u*)ap;
            ar[2 * t + 1] = *(const v4u*)(ap + 8);
        }
    };
    auto loadB = [&](int k0) {
        const int kw = k0 / NFEAT;            // uniform per chunk (192 = 6*32)
        const int cb = k0 % NFEAT;
        const int wp = bw + kw - 1;
        const bool ok = (wp >= 0) && (wp < WDIM);
        #pragma unroll
        for (int t = 0; t < 8; ++t) {
            int c = cb + (bkp0 + t) * 2;
            float v0 = 0.0f, v1 = 0.0f;
            if (ok) {
                const float* hp = &hsrc[bstr * b + cstr * c + wp];
                v0 = hp[0];
                v1 = hp[cstr];
            }
            bx[2 * t]     = v0;
            bx[2 * t + 1] = v1;
        }
    };
    auto drain = [&](int buf) {
        #pragma unroll
        for (int t = 0; t < 2; ++t) {
            *(v4u*)&At[buf][am0 + t * 128][ak0]     = ar[2 * t];
            *(v4u*)&At[buf][am0 + t * 128][ak0 + 8] = ar[2 * t + 1];
        }
        #pragma unroll
        for (int t = 0; t < 8; ++t)
            *(unsigned*)&Bt[buf][bw][(bkp0 + t) * 2] =
                pack2bf(bx[2 * t], bx[2 * t + 1]);
    };

    loadA(0); loadB(0); drain(0);
    __syncthreads();

    for (int chunk = 0; chunk < NK; ++chunk) {
        const int cur = chunk & 1;
        if (chunk + 1 < NK) {
            loadA((chunk + 1) * 32);
            loadB((chunk + 1) * 32);
            if (chunk + 2 < NK)
                __builtin_prefetch(
                    &UpkP[(size_t)(Mbase + am0) * KU + (chunk + 2) * 32 + ak0],
                    0, 3);
        }
        v16bf a0 = frag2(&At[cur][mo + lrow][kgA], &At[cur][mo + lrow][kgA + 16]);
        v16bf a1 = frag2(&At[cur][128 + mo + lrow][kgA],
                         &At[cur][128 + mo + lrow][kgA + 16]);
        #pragma unroll
        for (int nt = 0; nt < 8; ++nt) {
            const unsigned short* bp = &Bt[cur][nt * 16 + lrow][ksB];
            v16bf bf = frag2(bp, bp + 16);
            acc[nt] = __builtin_amdgcn_wmma_f32_16x16x32_bf16(
                false, a0, false, bf, (short)0, acc[nt], false, false);
            acc[8 + nt] = __builtin_amdgcn_wmma_f32_16x16x32_bf16(
                false, a1, false, bf, (short)0, acc[8 + nt], false, false);
        }
        if (chunk + 1 < NK) drain(cur ^ 1);
        __syncthreads();
    }

    // ---- gates (+i2s) -> LDS (bf16)
    const int hi = lane >> 4, col = lane & 15;
    #pragma unroll
    for (int half = 0; half < 2; ++half) {
        #pragma unroll
        for (int nt = 0; nt < 8; ++nt) {
            #pragma unroll
            for (int r = 0; r < 8; ++r) {
                int ml = half * 128 + mo + hi * 8 + r;   // block-local perm row
                int n = nt * 16 + col;
                float v = acc[half * 8 + nt][r] +
                          bf2f(i2s[(((size_t)b * HDIM + h) * G4 + Mbase + ml) *
                                       WDIM + n]);
                gL[ml][n] = f2bf(v);
            }
        }
    }
    __syncthreads();

    // ---- in-kernel LSTM pointwise update
    // thread -> feature fl (0..63), w-range [w0, w0+32)
    const int fl = tid >> 2;
    const int w0 = (tid & 3) * 32;
    const int f  = blockIdx.x * 64 + fl;           // global feature
    const size_t cbase = ((size_t)b * NFEAT + f) * WDIM + w0;
    const size_t obase = (((size_t)b * NFEAT + f) * HDIM + h) * WDIM + w0;
    const unsigned short* go_p = &gL[0 * 64 + fl][w0];
    const unsigned short* gf_p = &gL[1 * 64 + fl][w0];
    const unsigned short* gi_p = &gL[2 * 64 + fl][w0];
    const unsigned short* gg_p = &gL[3 * 64 + fl][w0];
    #pragma unroll
    for (int j = 0; j < 32; ++j) {
        float o  = hsig(bf2f(go_p[j]));
        float fg = hsig(bf2f(gf_p[j]));
        float ig = hsig(bf2f(gi_p[j]));
        float g  = tanhf(bf2f(gg_p[j]));
        float c  = fg * cst[cbase + j] + ig * g;
        float hn = o * tanhf(c);
        cst[cbase + j]    = c;
        out_hs[obase + j] = hn;
    }
}

// ---------------------------------------------------------------- launch glue
extern "C" void kernel_launch(void* const* d_in, const int* in_sizes, int n_in,
                              void* d_out, int out_size, void* d_ws, size_t ws_size,
                              hipStream_t stream) {
    (void)in_sizes; (void)n_in; (void)out_size; (void)ws_size;
    const float* x      = (const float*)d_in[0];
    const float* W_iof  = (const float*)d_in[1];
    const float* U_iof  = (const float*)d_in[2];
    const float* b_iof  = (const float*)d_in[3];
    const float* init_h = (const float*)d_in[4];
    const float* init_c = (const float*)d_in[5];
    float* out = (float*)d_out;

    // workspace layout (all 16B aligned)
    char* ws = (char*)d_ws;
    size_t szI2S = (size_t)BDIM * HDIM * G4 * WDIM * sizeof(unsigned short); // 402.7MB
    size_t szWm  = (size_t)G4 * CDIM * sizeof(unsigned short);
    size_t szU   = (size_t)G4 * KU * sizeof(unsigned short);
    unsigned short* i2s  = (unsigned short*)(ws);
    unsigned short* WmP  = (unsigned short*)(ws + szI2S);
    unsigned short* UpkP = (unsigned short*)(ws + szI2S + szWm);
    float*          cst  = (float*)(ws + szI2S + szWm + szU);

    prep_w_kernel<<<(G4 * CDIM + 255) / 256, 256, 0, stream>>>(W_iof, WmP);
    prep_u_kernel<<<(G4 * KU + 255) / 256, 256, 0, stream>>>(U_iof, UpkP);
    prep_c_kernel<<<(BDIM * NFEAT * WDIM + 255) / 256, 256, 0, stream>>>(init_c, cst);

    i2s_gemm_kernel<<<dim3(3, HDIM, BDIM), 256, 0, stream>>>(x, WmP, b_iof, i2s);

    for (int h = 0; h < HDIM; ++h) {
        rec_step_kernel<<<dim3(3, BDIM), 256, 0, stream>>>(UpkP, i2s, out,
                                                           init_h, cst, h);
    }
}
